// Attention_18743237280614
// MI455X (gfx1250) — matrix operand order repaired and checked by reading, not compiled, not run on hardware
//
#include <hip/hip_runtime.h>
#include <hip/hip_bf16.h>

typedef __attribute__((ext_vector_type(16))) _Float16 v16h;
typedef __attribute__((ext_vector_type(8)))  float    v8f;
typedef __attribute__((ext_vector_type(4)))  float    v4f;

#define S_LEN 512
#define D_SZ  96
#define W1S   112   // padded LDS row stride in halfs (224 B: 32B-aligned rows, low bank conflict)

// out[b,e] = softmax_s( sum_e sigmoid(seq@W1^T + b1 + tgt@W2^T + b2) , masked ) applied as
//            ctx = attn^T @ seq ; out = ctx @ W1^T + b1   (exact, since sum(attn)==1)
__global__ __launch_bounds__(256)
void attn_fused_kernel(const float* __restrict__ seq,
                       const float* __restrict__ tgt,
                       const unsigned char* __restrict__ mask,   // bool (B, S+1, 1), 1 byte/elem
                       const float* __restrict__ w1,
                       const float* __restrict__ b1,
                       const float* __restrict__ w2,
                       const float* __restrict__ b2,
                       float* __restrict__ out)
{
    __shared__ __align__(32) _Float16 w1h[D_SZ * W1S];  // W1 as f16, B-operand friendly
    __shared__ float combL[D_SZ];     // b1[e] + tgt_h[e]
    __shared__ float b1L[D_SZ];
    __shared__ float tgtL[D_SZ];
    __shared__ float scoresL[S_LEN];  // scores, then unnormalized exp(p)
    __shared__ float redL[256];       // softmax reductions
    __shared__ float partW[8 * D_SZ]; // per-wave ctx partials
    __shared__ float ctxL[D_SZ];

    const int t    = threadIdx.x;
    const int b    = blockIdx.x;
    const int lane = t & 31;
    const int wid  = t >> 5;
    const int col  = lane & 15;   // N within tile / M-row for A loads
    const int half = lane >> 4;   // lane-half per ISA WMMA layouts

    // ---- stage W1 (f32 -> f16) into LDS, row-major by e with padded stride ----
    for (int idx = t; idx < D_SZ * D_SZ; idx += 256) {
        int e = idx / D_SZ, d = idx - e * D_SZ;
        w1h[e * W1S + d] = (_Float16)w1[idx];
    }
    if (t < D_SZ) { tgtL[t] = tgt[(size_t)b * D_SZ + t]; b1L[t] = b1[t]; }
    __syncthreads();

    // ---- tgt_h[e] = tgt·w2[e,:] + b2[e]  (exact f32; tiny) ; comb = tgt_h + b1 ----
    if (t < D_SZ) {
        float s = b2[t];
        const float* wr = w2 + t * D_SZ;
        #pragma unroll 8
        for (int d = 0; d < D_SZ; ++d) s += tgtL[d] * wr[d];
        combL[t] = s + b1L[t];
    }
    __syncthreads();

    float combR[6];
    #pragma unroll
    for (int nt = 0; nt < 6; ++nt) combR[nt] = combL[nt * 16 + col];

    const size_t seqBase = (size_t)b * S_LEN * D_SZ;
    const unsigned char* mrow = mask + (size_t)b * (S_LEN + 1);

    // ================= Pass A: scores[s] via f16 WMMA =================
    // M = 512 rows (32 tiles of 16), N = 96 (6 tiles), K = 96 (3 chunks of 32)
    for (int mt = wid; mt < 32; mt += 8) {
        const int arow = mt * 16 + col;                       // A-row owned by this lane
        const float* rowp = seq + seqBase + (size_t)arow * D_SZ;
        if (mt + 8 < 32) __builtin_prefetch(rowp + 8 * 16 * D_SZ, 0, 0);

        // A operand: lane(row=col, half) holds K = 32c + half*16 + 0..15
        v16h a[3];
        #pragma unroll
        for (int c = 0; c < 3; ++c) {
            const float* g0 = rowp + c * 32 + half * 16;
            v4f f0 = *(const v4f*)(g0);
            v4f f1 = *(const v4f*)(g0 + 4);
            v4f f2 = *(const v4f*)(g0 + 8);
            v4f f3 = *(const v4f*)(g0 + 12);
            #pragma unroll
            for (int i = 0; i < 4; ++i) {
                a[c][i]      = (_Float16)f0[i];
                a[c][4 + i]  = (_Float16)f1[i];
                a[c][8 + i]  = (_Float16)f2[i];
                a[c][12 + i] = (_Float16)f3[i];
            }
        }

        v8f acc[6];
        #pragma unroll
        for (int nt = 0; nt < 6; ++nt) acc[nt] = (v8f){0.f,0.f,0.f,0.f,0.f,0.f,0.f,0.f};

        #pragma unroll
        for (int c = 0; c < 3; ++c) {
            #pragma unroll
            for (int nt = 0; nt < 6; ++nt) {
                // B operand: lane(n = nt*16+col, half) holds K = 32c + half*16 + 0..15
                const v16h bop =
                    *(const v16h*)&w1h[(nt * 16 + col) * W1S + c * 32 + half * 16];
                acc[nt] = __builtin_amdgcn_wmma_f32_16x16x32_f16(
                    false, a[c], false, bop, (short)0, acc[nt], false, false);
            }
        }

        // D layout: VGPR j of lane(col,half) = seq_h[row = mt*16 + half*8 + j][e = nt*16+col]
        #pragma unroll
        for (int j = 0; j < 8; ++j) {
            float sj = 0.f;
            #pragma unroll
            for (int nt = 0; nt < 6; ++nt) {
                float x = acc[nt][j] + combR[nt];
                sj += 1.0f / (1.0f + __expf(-x));   // sigmoid
            }
            // reduce the 96 e-terms: sum across the 16 lanes of this half-group
            #pragma unroll
            for (int m = 8; m >= 1; m >>= 1) sj += __shfl_xor(sj, m, 32);
            const int srow = mt * 16 + half * 8 + j;
            if (col == j)
                scoresL[srow] = mrow[srow] ? -1.0e9f : sj;
        }
    }
    __syncthreads();

    // ================= Softmax over s (block reduction in LDS) =================
    const float l0 = scoresL[t], l1 = scoresL[t + 256];
    redL[t] = fmaxf(l0, l1);
    __syncthreads();
    #pragma unroll
    for (int off = 128; off >= 1; off >>= 1) {
        if (t < off) redL[t] = fmaxf(redL[t], redL[t + off]);
        __syncthreads();
    }
    const float mx = redL[0];
    __syncthreads();
    const float p0 = __expf(l0 - mx), p1 = __expf(l1 - mx);
    scoresL[t] = p0; scoresL[t + 256] = p1;      // unnormalized weights
    redL[t] = p0 + p1;
    __syncthreads();
    #pragma unroll
    for (int off = 128; off >= 1; off >>= 1) {
        if (t < off) redL[t] += redL[t + off];
        __syncthreads();
    }
    const float inv = 1.0f / redL[0];
    __syncthreads();

    // ================= Pass B: ctx[d] = inv * sum_s p[s]*seq[b,s,d] =================
    // Each wave handles 64 s-rows; lane l accumulates d = l, l+32, l+64 (coalesced 384B/row).
    {
        float c0 = 0.f, c1 = 0.f, c2 = 0.f;
        const float* base = seq + seqBase + (size_t)(wid * 64) * D_SZ;
        for (int i = 0; i < 64; ++i) {
            const float p = scoresL[wid * 64 + i];
            const float* r = base + i * D_SZ;
            c0 += p * r[lane];
            c1 += p * r[lane + 32];
            c2 += p * r[lane + 64];
        }
        partW[wid * D_SZ + lane]      = c0;
        partW[wid * D_SZ + lane + 32] = c1;
        partW[wid * D_SZ + lane + 64] = c2;
    }
    __syncthreads();
    if (t < D_SZ) {
        float cx = 0.f;
        #pragma unroll
        for (int w = 0; w < 8; ++w) cx += partW[w * D_SZ + t];
        ctxL[t] = cx * inv;
    }
    __syncthreads();

    // ================= out[b,e] = ctx·w1[e,:] + b1[e]  (exact f32) =================
    if (t < D_SZ) {
        float o = b1L[t];
        const float* wr = w1 + t * D_SZ;
        #pragma unroll 8
        for (int d = 0; d < D_SZ; ++d) o += ctxL[d] * wr[d];
        out[(size_t)b * D_SZ + t] = o;
    }
}

extern "C" void kernel_launch(void* const* d_in, const int* in_sizes, int n_in,
                              void* d_out, int out_size, void* d_ws, size_t ws_size,
                              hipStream_t stream) {
    const float*         seq  = (const float*)d_in[0];
    const float*         tgt  = (const float*)d_in[1];
    const unsigned char* mask = (const unsigned char*)d_in[2];  // jax bool_: 1 byte/elem
    const float*         w1   = (const float*)d_in[3];
    const float*         b1   = (const float*)d_in[4];
    const float*         w2   = (const float*)d_in[5];
    const float*         b2   = (const float*)d_in[6];
    float*               out  = (float*)d_out;

    const int B = in_sizes[0] / (S_LEN * D_SZ);   // 2048
    attn_fused_kernel<<<dim3(B), dim3(256), 0, stream>>>(seq, tgt, mask, w1, b1, w2, b2, out);
    (void)n_in; (void)out_size; (void)d_ws; (void)ws_size;
}
